// EvaAdapterAttention_78400333021651
// MI455X (gfx1250) — matrix operand order repaired and checked by reading, not compiled, not run on hardware
//
#include <hip/hip_runtime.h>
#include <hip/hip_bf16.h>

// ---------- types ----------
typedef __attribute__((ext_vector_type(16))) _Float16 v16h;
typedef __attribute__((ext_vector_type(8)))  float    v8f;
typedef __attribute__((ext_vector_type(4)))  unsigned u32x4;
typedef __attribute__((ext_vector_type(4)))  float    f32x4;
typedef __attribute__((ext_vector_type(4)))  _Float16 h16x4;
typedef __attribute__((ext_vector_type(8)))  int      i32x8;
typedef __attribute__((ext_vector_type(4)))  int      i32x4;

union Frag { v16h v; u32x4 q[2]; };
union V8U  { u32x4 q; _Float16 h[8]; };

#define N_TOK 577
#define NTOT  18464      // B * N_TOK
#define NN    332929     // N_TOK * N_TOK

__device__ __forceinline__ v8f wmma16(v16h a, v16h b, v8f c) {
  return __builtin_amdgcn_wmma_f32_16x16x32_f16(false, a, false, b, (short)0, c, false, false);
}

// A fragment (16x32 f16) from row-major tile, rows m = lane&15, strideH in f16 (row base 16B aligned)
__device__ __forceinline__ v16h lds_frag_A(const _Float16* tile, int strideH) {
  int l = threadIdx.x & 31;
  int m = l & 15, kh = l >> 4;
  const char* row = (const char*)tile + (size_t)m * strideH * 2;
  Frag u;
  u.q[0] = *(const u32x4*)(row + kh * 16);
  u.q[1] = *(const u32x4*)(row + 32 + kh * 16);
  return u.v;
}

// B fragment (32x16 f16) where memory holds B^T rows (n-major, K contiguous)
__device__ __forceinline__ v16h lds_frag_B(const _Float16* tile, int strideH, int byteOff) {
  int l = threadIdx.x & 31;
  int n = l & 15, kh = l >> 4;
  const char* row = (const char*)tile + (size_t)n * strideH * 2 + byteOff;
  Frag u;
  u.q[0] = *(const u32x4*)(row + kh * 16);
  u.q[1] = *(const u32x4*)(row + 32 + kh * 16);
  return u.v;
}

// ---------- TDM: 2D tile (f16 elements) global -> LDS, rows padded 32 f16 -> stride 40 f16 ----------
// pad_interval=3 -> pad every 16 dwords (64B = 32 f16); pad_amount=3 -> 4 dwords (8 f16).
__device__ __forceinline__ void tdm_load_2d_f16(
    unsigned lds_byte_off, const void* gptr,
    unsigned tensor_d0, unsigned tensor_d1,   // remaining extent from tile start (elements); OOB -> 0
    unsigned tile_d0, unsigned tile_d1,       // tile (elements, rows)
    unsigned stride_d0)                       // row stride (elements)
{
  unsigned long long ga = (unsigned long long)gptr;
  u32x4 g0;
  g0[0] = 1u;                                  // count=1, user mode
  g0[1] = lds_byte_off;                        // lds_addr [63:32]
  g0[2] = (unsigned)ga;                        // global_addr lo
  g0[3] = (unsigned)(ga >> 32) | (2u << 30);   // global_addr hi | type=2 (image)
  i32x8 g1;
  g1[0] = (int)((1u << 16) | (1u << 20) | (3u << 22) | (3u << 25)); // data_size=2B, pad_en, interval, amount
  g1[1] = (int)((tensor_d0 & 0xFFFFu) << 16);                       // tensor_dim0 [79:48] lo
  g1[2] = (int)((tensor_d0 >> 16) | ((tensor_d1 & 0xFFFFu) << 16)); // dim0 hi | tensor_dim1 lo
  g1[3] = (int)(((tensor_d1 >> 16) & 0xFFFFu) | (tile_d0 << 16));   // dim1 hi | tile_dim0 [127:112]
  g1[4] = (int)(tile_d1 & 0xFFFFu);                                 // tile_dim1 [143:128], tile_dim2=0
  g1[5] = (int)stride_d0;                                           // tensor_dim0_stride [191:160]
  g1[6] = 0;                                                        // stride hi | dim1_stride lo
  g1[7] = 0;
  i32x4 g2 = {1, 0, 0, 0};  // tensor_dim2=1; tile_dim3=0 (unused)
  i32x4 g3 = {0, 0, 0, 0};
  i32x8 g4 = {0, 0, 0, 0, 0, 0, 0, 0};  // extra group required by 6-arg builtin (unused for 2D)
  __builtin_amdgcn_tensor_load_to_lds(g0, g1, g2, g3, g4, 0);
}

__device__ __forceinline__ unsigned lds_off_bytes(const void* p) {
  return (unsigned)(unsigned long long)p;   // low 32 bits of flat LDS address = LDS byte offset
}

// ---------- kernel: f32 -> f16 weight convert (vectorized x4) ----------
__global__ void cvt_f16_kernel(const float* __restrict__ in, _Float16* __restrict__ out, int n4) {
  int i = blockIdx.x * 256 + threadIdx.x;
  if (i >= n4) return;
  f32x4 v = ((const f32x4*)in)[i];
  h16x4 o;
  o[0] = (_Float16)v[0]; o[1] = (_Float16)v[1];
  o[2] = (_Float16)v[2]; o[3] = (_Float16)v[3];
  ((h16x4*)out)[i] = o;
}

// ---------- kernel: adapter (x + up(down(x))) -> f16 ----------
__global__ __launch_bounds__(256) void adapter_kernel(
    const float* __restrict__ x,  const float* __restrict__ aw, const float* __restrict__ ab,
    const float* __restrict__ bw, const float* __restrict__ bb, _Float16* __restrict__ xa) {
  __shared__ float wsum[8][9];
  __shared__ float down[8];
  const int tid = threadIdx.x;
  const int row = blockIdx.x;
  const int wid = tid >> 5, ln = tid & 31;
  f32x4 xv = ((const f32x4*)(x + (size_t)row * 1024))[tid];
  float part[8];
#pragma unroll
  for (int h = 0; h < 8; ++h) {
    f32x4 a = ((const f32x4*)(aw + h * 1024))[tid];
    float s = xv[0]*a[0] + xv[1]*a[1] + xv[2]*a[2] + xv[3]*a[3];
#pragma unroll
    for (int m = 16; m >= 1; m >>= 1) s += __shfl_xor(s, m, 32);
    part[h] = s;
  }
  if (ln == 0) {
#pragma unroll
    for (int h = 0; h < 8; ++h) wsum[wid][h] = part[h];
  }
  __syncthreads();
  if (tid < 8) {
    float s = ab[tid];
#pragma unroll
    for (int ww = 0; ww < 8; ++ww) s += wsum[ww][tid];
    down[tid] = s;
  }
  __syncthreads();
  h16x4 o4;
#pragma unroll
  for (int e = 0; e < 4; ++e) {
    int c = tid * 4 + e;
    float s = bb[c];
#pragma unroll
    for (int h = 0; h < 8; ++h) s += down[h] * bw[c * 8 + h];
    o4[e] = (_Float16)(xv[e] + s);
  }
  *(h16x4*)(xa + (size_t)row * 1024 + tid * 4) = o4;
}

// ---------- kernel: bias expand  table[idx[i,j], h] -> bias[h, i*577+j] ----------
__global__ void bias_expand_kernel(const float* __restrict__ table, const int* __restrict__ idx,
                                   float* __restrict__ outb) {
  int ij = blockIdx.x * 256 + threadIdx.x;
  if (ij >= NN) return;
  int id = idx[ij];
  const float* t = table + (size_t)id * 16;
#pragma unroll
  for (int h = 0; h < 16; ++h) outb[(size_t)h * NN + ij] = t[h];
}

// ---------- kernel: 128x128x32 WMMA GEMM with TDM double-buffered tiles ----------
// mode 0: QKV epilogue (scatter q/k/v f16, fold bias + 0.125 scale into q)
// mode 1: proj epilogue (f32 out + bias)
__global__ __launch_bounds__(256) void gemm128_kernel(
    const _Float16* __restrict__ A, const _Float16* __restrict__ W, int M, int mode,
    const float* __restrict__ qbias, const float* __restrict__ vbias,
    _Float16* __restrict__ qo, _Float16* __restrict__ ko, _Float16* __restrict__ vo,
    const float* __restrict__ pbias, float* __restrict__ outf) {
  __shared__ _Float16 At[2][128 * 40];
  __shared__ _Float16 Bt[2][128 * 40];
  const int tid = threadIdx.x;
  const int w = tid >> 5, l = tid & 31;
  const int n_ = l & 15, kh = l >> 4;
  const int m0 = blockIdx.x * 128, n0 = blockIdx.y * 128;
  const unsigned remM = (unsigned)(M - m0);       // rows remaining; TDM zero-fills beyond

  v8f acc[8];
#pragma unroll
  for (int j = 0; j < 8; ++j) acc[j] = (v8f){0,0,0,0,0,0,0,0};

  // prologue: wave 0 DMAs tile 0 (A and W) into buffer 0
  if (w == 0) {
    tdm_load_2d_f16(lds_off_bytes(&At[0][0]), A + (size_t)m0 * 1024,
                    1024u, remM, 32u, 128u, 1024u);
    tdm_load_2d_f16(lds_off_bytes(&Bt[0][0]), W + (size_t)n0 * 1024,
                    1024u, 128u, 32u, 128u, 1024u);
    __builtin_amdgcn_s_wait_tensorcnt(0);
  }
  __syncthreads();

  for (int s = 0; s < 32; ++s) {
    const int cur = s & 1, nxt = cur ^ 1;
    if (w == 0 && s + 1 < 32) {   // prefetch next K-slab while computing
      const unsigned k1 = (unsigned)(s + 1) * 32u;
      tdm_load_2d_f16(lds_off_bytes(&At[nxt][0]), A + (size_t)m0 * 1024 + k1,
                      1024u - k1, remM, 32u, 128u, 1024u);
      tdm_load_2d_f16(lds_off_bytes(&Bt[nxt][0]), W + (size_t)n0 * 1024 + k1,
                      1024u - k1, 128u, 32u, 128u, 1024u);
    }
    v16h a = lds_frag_A(&At[cur][0] + w * 16 * 40, 40);
#pragma unroll
    for (int j = 0; j < 8; ++j) {
      v16h b = lds_frag_B(&Bt[cur][0] + j * 16 * 40, 40, 0);
      acc[j] = wmma16(a, b, acc[j]);
    }
    __syncthreads();                                  // everyone done with cur
    if (s + 1 < 32) {
      if (w == 0) __builtin_amdgcn_s_wait_tensorcnt(0);
      __syncthreads();                                // nxt buffer ready for all waves
    }
  }

  int mrow[8], bIdx[8], tok[8];
  bool mval[8];
#pragma unroll
  for (int r = 0; r < 8; ++r) {
    int m = m0 + w * 16 + r + 8 * kh;
    mrow[r] = m; mval[r] = (m < M);
    int mm = mval[r] ? m : 0;
    bIdx[r] = mm / N_TOK;
    tok[r]  = mm - bIdx[r] * N_TOK;
  }
  if (mode == 0) {
#pragma unroll
    for (int j = 0; j < 8; ++j) {
      int o = n0 + j * 16 + n_;
      int which = o >> 10, c = o & 1023, h = c >> 6, d = c & 63;
      float bias = (which == 0) ? qbias[c] : ((which == 2) ? vbias[c] : 0.0f);
      float mult = (which == 0) ? 0.125f : 1.0f;
      _Float16* dst = (which == 0) ? qo : ((which == 1) ? ko : vo);
#pragma unroll
      for (int r = 0; r < 8; ++r) {
        if (!mval[r]) continue;
        float v = (acc[j][r] + bias) * mult;
        dst[((size_t)(bIdx[r] * 16 + h) * N_TOK + tok[r]) * 64 + d] = (_Float16)v;
      }
    }
  } else {
#pragma unroll
    for (int j = 0; j < 8; ++j) {
      int o = n0 + j * 16 + n_;
      float bias = pbias[o];
#pragma unroll
      for (int r = 0; r < 8; ++r)
        if (mval[r]) outf[(size_t)mrow[r] * 1024 + o] = acc[j][r] + bias;
    }
  }
}

// ---------- kernel: flash attention, per (b,h), 8 waves x 16 rows, 32-key chunks ----------
__global__ __launch_bounds__(256) void attn_kernel(
    const _Float16* __restrict__ q, const _Float16* __restrict__ k,
    const _Float16* __restrict__ v, const float* __restrict__ biasAll,
    _Float16* __restrict__ out) {
  __shared__ _Float16 Kt[32 * 72];       // K tile row-major [j][64], stride 72
  __shared__ _Float16 Vt[64 * 40];       // V tile transposed [d][32], stride 40
  __shared__ _Float16 Pt[8 * 16 * 40];   // per-wave P tiles [16][32], stride 40
  const int tid = threadIdx.x;
  const int w = tid >> 5, l = tid & 31, n_ = l & 15, kh = l >> 4;
  const int h = blockIdx.y, b = blockIdx.z;
  const int bh = b * 16 + h;
  const int i0 = blockIdx.x * 128 + w * 16;
  const float* biasrow = biasAll + (size_t)h * NN;
  _Float16* Pw = Pt + w * 16 * 40;

  int tq = i0 + (l & 15); if (tq > N_TOK - 1) tq = N_TOK - 1;
  const char* qrow = (const char*)(q + ((size_t)bh * N_TOK + tq) * 64);
  Frag fa0, fa1;
  fa0.q[0] = *(const u32x4*)(qrow + kh * 16);
  fa0.q[1] = *(const u32x4*)(qrow + 32 + kh * 16);
  fa1.q[0] = *(const u32x4*)(qrow + 64 + kh * 16);
  fa1.q[1] = *(const u32x4*)(qrow + 96 + kh * 16);

  int tokr[8];
#pragma unroll
  for (int r = 0; r < 8; ++r) { int t_ = i0 + r + 8 * kh; tokr[r] = (t_ > N_TOK - 1) ? N_TOK - 1 : t_; }

  float mrun[8], lrun[8];
  v8f o[4];
#pragma unroll
  for (int r = 0; r < 8; ++r) { mrun[r] = -1e30f; lrun[r] = 0.0f; }
#pragma unroll
  for (int t = 0; t < 4; ++t) o[t] = (v8f){0,0,0,0,0,0,0,0};

  for (int j0 = 0; j0 < N_TOK; j0 += 32) {
    __syncthreads();
    {  // cooperative K/V tile load; V stored transposed
      int jj = tid >> 3, c = tid & 7;
      int j = j0 + jj;
      u32x4 kv = {0, 0, 0, 0}, vv = {0, 0, 0, 0};
      if (j < N_TOK) {
        kv = *(const u32x4*)(k + ((size_t)bh * N_TOK + j) * 64 + c * 8);
        vv = *(const u32x4*)(v + ((size_t)bh * N_TOK + j) * 64 + c * 8);
      }
      *(u32x4*)((char*)Kt + jj * 144 + c * 16) = kv;
      V8U uv; uv.q = vv;
#pragma unroll
      for (int e = 0; e < 8; ++e) Vt[(c * 8 + e) * 40 + jj] = uv.h[e];
    }
    __syncthreads();

    v8f s0 = (v8f){0,0,0,0,0,0,0,0}, s1 = (v8f){0,0,0,0,0,0,0,0};
    {
      v16h kb;
      kb = lds_frag_B(Kt, 72, 0);            s0 = wmma16(fa0.v, kb, s0);
      kb = lds_frag_B(Kt, 72, 64);           s0 = wmma16(fa1.v, kb, s0);
      kb = lds_frag_B(Kt + 16 * 72, 72, 0);  s1 = wmma16(fa0.v, kb, s1);
      kb = lds_frag_B(Kt + 16 * 72, 72, 64); s1 = wmma16(fa1.v, kb, s1);
    }

    int jA = j0 + n_, jB = j0 + 16 + n_;
    bool vA = jA < N_TOK, vB = jB < N_TOK;
#pragma unroll
    for (int r = 0; r < 8; ++r) {
      float x0 = vA ? (s0[r] + biasrow[(size_t)tokr[r] * N_TOK + jA]) : -1e30f;
      float x1 = vB ? (s1[r] + biasrow[(size_t)tokr[r] * N_TOK + jB]) : -1e30f;
      float rm = fmaxf(x0, x1);
#pragma unroll
      for (int msk = 8; msk >= 1; msk >>= 1) rm = fmaxf(rm, __shfl_xor(rm, msk, 16));
      float mn = fmaxf(mrun[r], rm);
      const float L2E = 1.4426950408889634f;
      float al = exp2f((mrun[r] - mn) * L2E);
      float p0 = exp2f((x0 - mn) * L2E);
      float p1 = exp2f((x1 - mn) * L2E);
      float rs = p0 + p1;
#pragma unroll
      for (int msk = 8; msk >= 1; msk >>= 1) rs += __shfl_xor(rs, msk, 16);
      lrun[r] = lrun[r] * al + rs;
      mrun[r] = mn;
#pragma unroll
      for (int t = 0; t < 4; ++t) o[t][r] *= al;
      Pw[(r + 8 * kh) * 40 + n_]      = (_Float16)p0;
      Pw[(r + 8 * kh) * 40 + 16 + n_] = (_Float16)p1;
    }
    asm volatile("s_wait_dscnt 0" ::: "memory");  // P store -> A-frag read (same wave)
    v16h pa = lds_frag_A(Pw, 40);
#pragma unroll
    for (int t = 0; t < 4; ++t) {
      v16h vb_ = lds_frag_B(Vt + t * 16 * 40, 40, 0);
      o[t] = wmma16(pa, vb_, o[t]);
    }
  }

#pragma unroll
  for (int r = 0; r < 8; ++r) {
    int t_ = i0 + r + 8 * kh;
    if (t_ >= N_TOK) continue;
    float inv = 1.0f / lrun[r];
    _Float16* dst = out + ((size_t)b * N_TOK + t_) * 1024 + h * 64;
#pragma unroll
    for (int t = 0; t < 4; ++t) dst[t * 16 + n_] = (_Float16)(o[t][r] * inv);
  }
}

// ---------- launcher ----------
extern "C" void kernel_launch(void* const* d_in, const int* in_sizes, int n_in,
                              void* d_out, int out_size, void* d_ws, size_t ws_size,
                              hipStream_t stream) {
  const float* x     = (const float*)d_in[0];
  const float* aw    = (const float*)d_in[1];
  const float* ab    = (const float*)d_in[2];
  const float* bw    = (const float*)d_in[3];
  const float* bb    = (const float*)d_in[4];
  const float* qkvw  = (const float*)d_in[5];
  const float* qbias = (const float*)d_in[6];
  const float* vbias = (const float*)d_in[7];
  const float* table = (const float*)d_in[8];
  const float* projw = (const float*)d_in[9];
  const float* projb = (const float*)d_in[10];
  const int*   rpi   = (const int*)d_in[11];
  float* out = (float*)d_out;

  char* ws = (char*)d_ws;
  _Float16* xa    = (_Float16*)(ws + 0);            // 37,814,272 (also attention output, aliased)
  _Float16* wq16  = (_Float16*)(ws + 37814272ull);  //  6,291,456
  _Float16* wp16  = (_Float16*)(ws + 44105728ull);  //  2,097,152
  _Float16* qbuf  = (_Float16*)(ws + 46202880ull);  // 37,814,272
  _Float16* kbuf  = (_Float16*)(ws + 84017152ull);  // 37,814,272
  _Float16* vbuf  = (_Float16*)(ws + 121831424ull); // 37,814,272
  float*    biasb = (float*)   (ws + 159645696ull); // 21,307,456 -> total 180,953,152

  cvt_f16_kernel<<<3072, 256, 0, stream>>>(qkvw, wq16, 786432);
  cvt_f16_kernel<<<1024, 256, 0, stream>>>(projw, wp16, 262144);
  adapter_kernel<<<NTOT, 256, 0, stream>>>(x, aw, ab, bw, bb, xa);
  gemm128_kernel<<<dim3(145, 24), 256, 0, stream>>>(xa, wq16, NTOT, 0,
      qbias, vbias, qbuf, kbuf, vbuf, nullptr, nullptr);
  bias_expand_kernel<<<1301, 256, 0, stream>>>(table, rpi, biasb);
  attn_kernel<<<dim3(5, 16, 32), 256, 0, stream>>>(qbuf, kbuf, vbuf, biasb, xa);
  gemm128_kernel<<<dim3(145, 8), 256, 0, stream>>>(xa, wp16, NTOT, 1,
      nullptr, nullptr, nullptr, nullptr, nullptr, projb, out);
}
